// TRMAttentionLowRank_11416023072958
// MI455X (gfx1250) — matrix operand order repaired and checked by reading, not compiled, not run on hardware
//
#include <hip/hip_runtime.h>

typedef __attribute__((ext_vector_type(16))) _Float16 v16h;
typedef __attribute__((ext_vector_type(8)))  _Float16 v8h;
typedef __attribute__((ext_vector_type(4)))  _Float16 v4h;
typedef __attribute__((ext_vector_type(8)))  float    v8f;

#define B_   16
#define L_   2048
#define HID_ 1024
#define H_   16
#define DH_  64
#define K_   16
#define ML   (B_*L_)        // 32768 rows total

// ---------------------------------------------------------------------------
// CDNA5 async copy: global -> LDS, 16 bytes per lane, tracked by ASYNCcnt.
// ---------------------------------------------------------------------------
__device__ __forceinline__ void async_copy_b128(unsigned lds_addr, unsigned long long gaddr)
{
  asm volatile("global_load_async_to_lds_b128 %0, %1, off"
               :: "v"(lds_addr), "v"(gaddr) : "memory");
}
__device__ __forceinline__ void wait_asynccnt0()
{
  asm volatile("s_wait_asynccnt 0x0" ::: "memory");
}

// ---------------------------------------------------------------------------
// WMMA GEMM:  C[M,N] = A16[M,Kd] * W16[N,Kd]^T   (f16 in, f32 accumulate/out)
// Block tile 128x128, K-step 32, 8 wave32 waves (4 along M x 2 along N),
// double-buffered LDS staged by global_load_async_to_lds_b128.
// ---------------------------------------------------------------------------
#define TM  128
#define TN  128
#define TKK 32
#define LDT 40     // padded row stride in halves (80 bytes, 16B aligned)

__global__ __launch_bounds__(256) void wmma_gemm_kernel(
    const _Float16* __restrict__ A16, const _Float16* __restrict__ W16,
    float* __restrict__ C, int M, int N, int Kd)
{
  __shared__ _Float16 sA[2][TM * LDT];
  __shared__ _Float16 sB[2][TN * LDT];

  const int tid   = threadIdx.x;
  const int lane  = tid & 31;
  const int wave  = tid >> 5;
  const int wm    = wave & 3;     // wave position along M (0..3) -> 32 rows
  const int wn    = wave >> 2;    // wave position along N (0..1) -> 64 cols
  const int m0g   = blockIdx.y * TM;
  const int n0g   = blockIdx.x * TN;
  const int col16 = lane & 15;
  const int hi    = lane >> 4;

  // staging geometry: 512 x 16B slots per tile, 2 slots per thread
  const int srow = tid >> 2;            // rows 0..63 (slot0), +64 (slot1)
  const int sc8  = (tid & 3) << 3;      // half offset within row: 0,8,16,24
  unsigned long long gA = (unsigned long long)(uintptr_t)(A16 + (size_t)(m0g + srow) * Kd + sc8);
  unsigned long long gB = (unsigned long long)(uintptr_t)(W16 + (size_t)(n0g + srow) * Kd + sc8);
  const unsigned long long gRow = (unsigned long long)64 * Kd * 2;   // slot1 byte offset
  unsigned lA = (unsigned)(uintptr_t)&sA[0][srow * LDT + sc8];
  unsigned lB = (unsigned)(uintptr_t)&sB[0][srow * LDT + sc8];
  const unsigned lRow = 64u * LDT * 2u;          // slot1 LDS byte offset
  const unsigned lBuf = (unsigned)(TM * LDT * 2); // ping-pong buffer stride

  v8f c[2][4];
  #pragma unroll
  for (int i = 0; i < 2; ++i)
    #pragma unroll
    for (int j = 0; j < 4; ++j)
      #pragma unroll
      for (int e = 0; e < 8; ++e) c[i][j][e] = 0.0f;

  const int nIter = Kd / TKK;

  // prologue: stage tile 0 into buffer 0
  async_copy_b128(lA,        gA);
  async_copy_b128(lA + lRow, gA + gRow);
  async_copy_b128(lB,        gB);
  async_copy_b128(lB + lRow, gB + gRow);
  wait_asynccnt0();
  __syncthreads();

  for (int it = 0; it < nIter; ++it) {
    const int buf = it & 1;

    // stage tile it+1 into the other buffer (async, no wait yet)
    if (it + 1 < nIter) {
      unsigned long long go = (unsigned long long)((it + 1) * TKK) * 2u;
      unsigned lb = (buf ^ 1) * lBuf;
      async_copy_b128(lA + lb,        gA + go);
      async_copy_b128(lA + lb + lRow, gA + go + gRow);
      async_copy_b128(lB + lb,        gB + go);
      async_copy_b128(lB + lb + lRow, gB + go + gRow);
    }

    // ---- fragments per ISA VGPR layouts, from current buffer ----
    v16h a[2], bb[4];
    #pragma unroll
    for (int i = 0; i < 2; ++i) {
      int r = wm * 32 + i * 16 + col16;
      v8h lo  = *(const v8h*)&sA[buf][r * LDT + hi * 8];
      v8h hi8 = *(const v8h*)&sA[buf][r * LDT + hi * 8 + 16];
      a[i] = __builtin_shufflevector(lo, hi8, 0,1,2,3,4,5,6,7,8,9,10,11,12,13,14,15);
    }
    #pragma unroll
    for (int j = 0; j < 4; ++j) {
      int n = wn * 64 + j * 16 + col16;
      v8h lo  = *(const v8h*)&sB[buf][n * LDT + hi * 8];
      v8h hi8 = *(const v8h*)&sB[buf][n * LDT + hi * 8 + 16];
      bb[j] = __builtin_shufflevector(lo, hi8, 0,1,2,3,4,5,6,7,8,9,10,11,12,13,14,15);
    }
    #pragma unroll
    for (int i = 0; i < 2; ++i)
      #pragma unroll
      for (int j = 0; j < 4; ++j)
        c[i][j] = __builtin_amdgcn_wmma_f32_16x16x32_f16(
            false, a[i], false, bb[j], (short)0, c[i][j], false, false);

    if (it + 1 < nIter) wait_asynccnt0();
    __syncthreads();
  }

  // ---- store: C layout VGPR r -> M = r + 8*(lane>=16), N = lane&15 ----
  #pragma unroll
  for (int i = 0; i < 2; ++i)
    #pragma unroll
    for (int j = 0; j < 4; ++j) {
      int rb = m0g + wm * 32 + i * 16 + hi * 8;
      int cb = n0g + wn * 64 + j * 16 + col16;
      #pragma unroll
      for (int r = 0; r < 8; ++r)
        C[(size_t)(rb + r) * N + cb] = c[i][j][r];
    }
}

// ---------------------------------------------------------------------------
// fp32 -> f16 conversion (vectorized), n4 = count/4
// ---------------------------------------------------------------------------
__global__ __launch_bounds__(256) void cvt_f16_kernel(
    const float* __restrict__ in, _Float16* __restrict__ out, int n4)
{
  int t = blockIdx.x * 256 + threadIdx.x;
  if (t < n4) {
    float4 f = ((const float4*)in)[t];
    v4h h;
    h[0] = (_Float16)f.x; h[1] = (_Float16)f.y;
    h[2] = (_Float16)f.z; h[3] = (_Float16)f.w;
    *(v4h*)(out + 4 * (size_t)t) = h;
  }
}

// ---------------------------------------------------------------------------
// valid[b,l] = (attn_mask[b,0,l,l] == 0)
// ---------------------------------------------------------------------------
__global__ void init_valid_kernel(const float* __restrict__ mask, float* __restrict__ valid)
{
  int t = blockIdx.x * blockDim.x + threadIdx.x;
  int b = t >> 11, l = t & (L_ - 1);
  float v = mask[(size_t)b * L_ * L_ + (size_t)l * L_ + l];
  valid[t] = (v == 0.0f) ? 1.0f : 0.0f;
}

// ---------------------------------------------------------------------------
// D[ml,k] = sum_d X[ml,d] * theta[d,k]
// ---------------------------------------------------------------------------
__global__ __launch_bounds__(256) void compute_D_kernel(
    const float* __restrict__ X, const float* __restrict__ theta, float* __restrict__ D)
{
  __shared__ float sX[16][64];
  __shared__ float sT[64][16];
  int tid = threadIdx.x;
  int l0  = blockIdx.x * 16;
  int ll = tid >> 4, kk = tid & 15;
  float acc = 0.0f;
  for (int d0 = 0; d0 < HID_; d0 += 64) {
    #pragma unroll
    for (int i = 0; i < 4; ++i) {
      int e = tid + i * 256;
      sX[e >> 6][e & 63] = X[(size_t)(l0 + (e >> 6)) * HID_ + d0 + (e & 63)];
      sT[e >> 4][e & 15] = theta[(size_t)(d0 + (e >> 4)) * K_ + (e & 15)];
    }
    __syncthreads();
    #pragma unroll
    for (int d = 0; d < 64; ++d) acc = fmaf(sX[ll][d], sT[d][kk], acc);
    __syncthreads();
  }
  D[(size_t)(l0 + ll) * K_ + kk] = acc;
}

// ---------------------------------------------------------------------------
// Masked softmax over L per (b,k), renormalized over valid rows.
// ---------------------------------------------------------------------------
__global__ __launch_bounds__(256) void softmax_D_kernel(
    float* __restrict__ D, const float* __restrict__ valid)
{
  __shared__ float red[256];
  int tid = threadIdx.x;
  int b = blockIdx.x >> 4;
  int k = blockIdx.x & 15;
  float x[8], v[8];
  float m = -3.0e38f;
  #pragma unroll
  for (int j = 0; j < 8; ++j) {
    int l = tid + j * 256;
    x[j] = D[(size_t)(b * L_ + l) * K_ + k];
    v[j] = valid[b * L_ + l];
    if (v[j] > 0.5f) m = fmaxf(m, x[j]);
  }
  red[tid] = m; __syncthreads();
  for (int s = 128; s > 0; s >>= 1) { if (tid < s) red[tid] = fmaxf(red[tid], red[tid + s]); __syncthreads(); }
  float Mx = red[0]; __syncthreads();
  float e[8], ssum = 0.0f;
  #pragma unroll
  for (int j = 0; j < 8; ++j) {
    e[j] = (v[j] > 0.5f) ? __expf(x[j] - Mx) : 0.0f;
    ssum += e[j];
  }
  red[tid] = ssum; __syncthreads();
  for (int s = 128; s > 0; s >>= 1) { if (tid < s) red[tid] += red[tid + s]; __syncthreads(); }
  float inv = 1.0f / fmaxf(red[0], 1e-9f);
  #pragma unroll
  for (int j = 0; j < 8; ++j) {
    int l = tid + j * 256;
    D[(size_t)(b * L_ + l) * K_ + k] = e[j] * inv;
  }
}

// ---------------------------------------------------------------------------
// agg[b,k,d] = sum_l w[b,l,k] * X[b,l,d]   (K and V fused)
// ---------------------------------------------------------------------------
__global__ __launch_bounds__(256) void aggregate_kernel(
    const float* __restrict__ Kf, const float* __restrict__ Vf,
    const float* __restrict__ Dk, const float* __restrict__ Dv,
    float* __restrict__ kagg, float* __restrict__ vagg)
{
  __shared__ float sWk[128][K_];
  __shared__ float sWv[128][K_];
  int tid = threadIdx.x;
  int b = blockIdx.y;
  int d = blockIdx.x * 256 + tid;
  float ak[K_], av[K_];
  #pragma unroll
  for (int k = 0; k < K_; ++k) { ak[k] = 0.0f; av[k] = 0.0f; }
  for (int l0 = 0; l0 < L_; l0 += 128) {
    #pragma unroll
    for (int i = 0; i < 8; ++i) {
      int e = tid + i * 256;
      sWk[e >> 4][e & 15] = Dk[(size_t)(b * L_ + l0 + (e >> 4)) * K_ + (e & 15)];
      sWv[e >> 4][e & 15] = Dv[(size_t)(b * L_ + l0 + (e >> 4)) * K_ + (e & 15)];
    }
    __syncthreads();
    for (int l = 0; l < 128; ++l) {
      float kx = Kf[(size_t)(b * L_ + l0 + l) * HID_ + d];
      float vx = Vf[(size_t)(b * L_ + l0 + l) * HID_ + d];
      #pragma unroll
      for (int k = 0; k < K_; ++k) {
        ak[k] = fmaf(sWk[l][k], kx, ak[k]);
        av[k] = fmaf(sWv[l][k], vx, av[k]);
      }
    }
    __syncthreads();
  }
  #pragma unroll
  for (int k = 0; k < K_; ++k) {
    kagg[(size_t)(b * K_ + k) * HID_ + d] = ak[k];
    vagg[(size_t)(b * K_ + k) * HID_ + d] = av[k];
  }
}

// ---------------------------------------------------------------------------
// Fused RoPE + 16-slot attention; reads Q (fp32), writes AO (f16) for the
// final WMMA GEMM.
// ---------------------------------------------------------------------------
__global__ __launch_bounds__(256) void attention_kernel(
    const float* __restrict__ Q, _Float16* __restrict__ AO,
    const float* __restrict__ kagg, const float* __restrict__ vagg,
    const float* __restrict__ cosb, const float* __restrict__ sinb,
    const float* __restrict__ valid)
{
  __shared__ float sK[K_][DH_];
  __shared__ float sV[K_][DH_];
  int tid = threadIdx.x;
  int b = blockIdx.z, h = blockIdx.y;
  int l = blockIdx.x * 256 + tid;
  #pragma unroll
  for (int i = 0; i < 4; ++i) {
    int e = tid + i * 256;
    int k = e >> 6, j = e & 63;
    sK[k][j] = kagg[(size_t)(b * K_ + k) * HID_ + h * DH_ + j];
    sV[k][j] = vagg[(size_t)(b * K_ + k) * HID_ + h * DH_ + j];
  }
  __syncthreads();

  size_t base = (size_t)(b * L_ + l) * HID_ + h * DH_;
  float q[DH_];
  const float4* qp = (const float4*)(Q + base);
  #pragma unroll
  for (int i = 0; i < 16; ++i) {
    float4 f = qp[i];
    q[4*i] = f.x; q[4*i+1] = f.y; q[4*i+2] = f.z; q[4*i+3] = f.w;
  }
  // RoPE: q' = q*cos + rotate_half(q)*sin, rotate_half = [-x2, x1]
  #pragma unroll
  for (int j = 0; j < 32; ++j) {
    float cj  = cosb[l * DH_ + j],      sj  = sinb[l * DH_ + j];
    float cj2 = cosb[l * DH_ + j + 32], sj2 = sinb[l * DH_ + j + 32];
    float a0 = q[j], a1 = q[j + 32];
    q[j]      = a0 * cj  - a1 * sj;
    q[j + 32] = a1 * cj2 + a0 * sj2;
  }
  bool pad = valid[b * L_ + l] < 0.5f;
  float sc[K_];
  #pragma unroll
  for (int k = 0; k < K_; ++k) {
    float s = 0.0f;
    #pragma unroll
    for (int j = 0; j < DH_; ++j) s = fmaf(q[j], sK[k][j], s);
    sc[k] = pad ? 0.0f : s * 0.125f;   // pad rows -> uniform softmax (matches ref)
  }
  float m = sc[0];
  #pragma unroll
  for (int k = 1; k < K_; ++k) m = fmaxf(m, sc[k]);
  float ssum = 0.0f;
  #pragma unroll
  for (int k = 0; k < K_; ++k) { sc[k] = __expf(sc[k] - m); ssum += sc[k]; }
  float inv = 1.0f / ssum;

  #pragma unroll
  for (int i = 0; i < 16; ++i) {
    float o0 = 0.f, o1 = 0.f, o2 = 0.f, o3 = 0.f;
    #pragma unroll
    for (int k = 0; k < K_; ++k) {
      float p = sc[k];
      o0 = fmaf(p, sV[k][4*i],     o0);
      o1 = fmaf(p, sV[k][4*i + 1], o1);
      o2 = fmaf(p, sV[k][4*i + 2], o2);
      o3 = fmaf(p, sV[k][4*i + 3], o3);
    }
    v4h h4;
    h4[0] = (_Float16)(o0 * inv); h4[1] = (_Float16)(o1 * inv);
    h4[2] = (_Float16)(o2 * inv); h4[3] = (_Float16)(o3 * inv);
    *(v4h*)(AO + base + 4 * i) = h4;
  }
}

// ---------------------------------------------------------------------------
extern "C" void kernel_launch(void* const* d_in, const int* in_sizes, int n_in,
                              void* d_out, int out_size, void* d_ws, size_t ws_size,
                              hipStream_t stream)
{
  (void)in_sizes; (void)n_in; (void)out_size; (void)ws_size;
  const float* hs   = (const float*)d_in[0];
  const float* cosb = (const float*)d_in[1];
  const float* sinb = (const float*)d_in[2];
  const float* mask = (const float*)d_in[3];
  const float* Wq   = (const float*)d_in[4];
  const float* Wk   = (const float*)d_in[5];
  const float* Wv   = (const float*)d_in[6];
  const float* Wo   = (const float*)d_in[7];
  const float* thk  = (const float*)d_in[8];
  const float* thv  = (const float*)d_in[9];
  float* out = (float*)d_out;
  float* ws  = (float*)d_ws;

  // ---- workspace layout (~417 MB) ----
  float* Q   = ws;                                  // 32768*1024 f32
  float* Kf  = Q   + (size_t)ML * HID_;             // 32768*1024 f32
  float* Dk  = Kf  + (size_t)ML * HID_;             // 32768*16
  float* Dv  = Dk  + (size_t)ML * K_;
  float* vld = Dv  + (size_t)ML * K_;               // 32768
  float* kag = vld + ML;                            // 16*16*1024
  float* vag = kag + (size_t)B_ * K_ * HID_;
  _Float16* hs16 = (_Float16*)(vag + (size_t)B_ * K_ * HID_);   // 32768*1024 f16
  _Float16* w16  = hs16 + (size_t)ML * HID_;                    // 4*1024*1024 f16
  _Float16* ao16 = w16  + (size_t)4 * HID_ * HID_;              // 32768*1024 f16
  _Float16* wq16 = w16;
  _Float16* wk16 = w16 + (size_t)1 * HID_ * HID_;
  _Float16* wv16 = w16 + (size_t)2 * HID_ * HID_;
  _Float16* wo16 = w16 + (size_t)3 * HID_ * HID_;
  float* Vf = out;   // reuse d_out as V_full scratch; overwritten by final GEMM

  // ---- pre-convert activations + weights to f16 (hs16 fits in 192MB L2) ----
  cvt_f16_kernel<<<(ML * HID_ / 4) / 256, 256, 0, stream>>>(hs, hs16, ML * HID_ / 4);
  cvt_f16_kernel<<<(HID_ * HID_ / 4) / 256, 256, 0, stream>>>(Wq, wq16, HID_ * HID_ / 4);
  cvt_f16_kernel<<<(HID_ * HID_ / 4) / 256, 256, 0, stream>>>(Wk, wk16, HID_ * HID_ / 4);
  cvt_f16_kernel<<<(HID_ * HID_ / 4) / 256, 256, 0, stream>>>(Wv, wv16, HID_ * HID_ / 4);
  cvt_f16_kernel<<<(HID_ * HID_ / 4) / 256, 256, 0, stream>>>(Wo, wo16, HID_ * HID_ / 4);

  init_valid_kernel<<<ML / 256, 256, 0, stream>>>(mask, vld);

  dim3 gg(HID_ / TN, ML / TM);
  wmma_gemm_kernel<<<gg, 256, 0, stream>>>(hs16, wq16, Q,  ML, HID_, HID_);
  wmma_gemm_kernel<<<gg, 256, 0, stream>>>(hs16, wk16, Kf, ML, HID_, HID_);
  wmma_gemm_kernel<<<gg, 256, 0, stream>>>(hs16, wv16, Vf, ML, HID_, HID_);

  compute_D_kernel<<<ML / 16, 256, 0, stream>>>(Kf, thk, Dk);
  compute_D_kernel<<<ML / 16, 256, 0, stream>>>(Vf, thv, Dv);

  softmax_D_kernel<<<B_ * K_, 256, 0, stream>>>(Dk, vld);
  softmax_D_kernel<<<B_ * K_, 256, 0, stream>>>(Dv, vld);

  aggregate_kernel<<<dim3(HID_ / 256, B_), 256, 0, stream>>>(Kf, Vf, Dk, Dv, kag, vag);

  attention_kernel<<<dim3(L_ / 256, H_, B_), 256, 0, stream>>>(Q, ao16, kag, vag, cosb, sinb, vld);

  wmma_gemm_kernel<<<gg, 256, 0, stream>>>(ao16, wo16, out, ML, HID_, HID_);
}
